// PolicyNet_83073257439404
// MI455X (gfx1250) — matrix operand order repaired and checked by reading, not compiled, not run on hardware
//
#include <hip/hip_runtime.h>
#include <hip/hip_bf16.h>
#include <math.h>

// ---------------- problem constants ----------------
#define BATCH   16384
#define STATE   150
#define SPAD    160      // STATE padded to multiple of 32 (WMMA K tile)
#define HID     64
#define ACT     10
#define TEAM    5
#define STEPS   5
#define NGATES  256
#define QKVD    192

typedef __attribute__((ext_vector_type(16))) _Float16 v16h;
typedef __attribute__((ext_vector_type(8)))  float    v8f;

struct Params {
  const float *x;
  const float *fc1_w, *fc1_b, *fc2_w, *fc2_b, *fc3_w, *fc3_b;
  const float *h0_w, *h0_b, *c0_w, *c0_b;
  const float *lstm_whh, *lstm_bih, *lstm_bhh;   // lstm_wih unused (input is zeros)
  const float *predfc_w, *predfc_b, *emb_w, *emb_b;
  const float *attn_in_w, *attn_in_b, *attn_out_w, *attn_out_b;
  const float *unemb_w, *unemb_b;
  float *out_probs;   // [BATCH, ACT]
  float *out_pred;    // [BATCH, STEPS, TEAM, STATE]
};

__device__ __forceinline__ float sigm(float x) { return 1.0f / (1.0f + __expf(-x)); }

// ---- A-tile pack: 16x32 f16 from LDS (row-major, lda halfs), per ISA 16-bit A layout
__device__ __forceinline__ v16h load_A(const _Float16* src, int lda, int k0, int lane) {
  const int half_id = lane >> 4;
  const int m = lane & 15;
  const _Float16* row = src + m * lda + k0 + half_id * 8;
  v16h a;
#pragma unroll
  for (int j = 0; j < 8; ++j) {
    const int kb = ((j & 4) << 2) + ((j & 3) << 1);   // 0,2,4,6,16,18,20,22
    a[2 * j]     = row[kb];
    a[2 * j + 1] = row[kb + 1];
  }
  return a;
}

// ---- B-tile pack: 32x16 (K x N) from row-major f32 W[N][K].
// BRANCHLESS: clamp indices to valid addresses (loads always issue, no exec
// masking) and zero-select values with v_cndmask. K must be even (64, 150 are).
__device__ __forceinline__ v16h load_B(const float* W, int K, int N, int n0, int k0, int lane) {
  const int half_id = lane >> 4;
  const int nr = (lane & 15) + n0;
  const bool nok = (nr < N);
  const int n = nok ? nr : (N - 1);                 // clamped, always valid row
  const float2* W2 = (const float2*)(W + (size_t)n * K);
  v16h b;
#pragma unroll
  for (int j = 0; j < 8; ++j) {
    const int k = k0 + half_id * 8 + ((j & 4) << 2) + ((j & 3) << 1);
    const bool kok = (k + 1) < K;                   // K even => pair-aligned
    const float2 w = W2[kok ? (k >> 1) : 0];        // clamped, always in-bounds
    const bool ok = nok && kok;
    b[2 * j]     = ok ? (_Float16)w.x : (_Float16)0.0f;
    b[2 * j + 1] = ok ? (_Float16)w.y : (_Float16)0.0f;
  }
  return b;
}

// ---- Generic 16xN = 16xKpad @ Wt GEMM on the WMMA pipe; store via callback per 16-col tile.
template <typename StoreFn>
__device__ __forceinline__ void gemm_tile(const _Float16* A, int lda,
                                          const float* W, int K, int Kpad, int N,
                                          int lane, StoreFn store) {
  const int nTiles = (N + 15) >> 4;
  const int kTiles = Kpad >> 5;
  for (int ni = 0; ni < nTiles; ++ni) {
    v8f acc = {};
    for (int ki = 0; ki < kTiles; ++ki) {
      v16h a = load_A(A, lda, ki << 5, lane);
      v16h b = load_B(W, K, N, ni << 4, ki << 5, lane);
      acc = __builtin_amdgcn_wmma_f32_16x16x32_f16(false, a, false, b,
                                                   (short)0, acc, false, false);
    }
    store(ni << 4, acc);
  }
}

__global__ __launch_bounds__(32) void policy_fused(Params p) {
  // Persistent per-tile state
  __shared__ __align__(16) _Float16 sH2[16 * HID];    //  2 KB
  __shared__ __align__(16) _Float16 sH [16 * HID];    //  2 KB
  __shared__ __align__(16) float    sC [16 * HID];    //  4 KB
  __shared__ __align__(16) _Float16 sE [80 * HID];    // 10 KB  (embeds / proj result)
  // Union scratch: x-tile+h1 | logits f32 | gates f32 | pred_step | qkv + attn-out
  __shared__ __align__(16) _Float16 sBig[20480];      // 40 KB

  const int lane = threadIdx.x;
  const int b0   = blockIdx.x * 16;

  _Float16* sX     = sBig;                  // 16 x SPAD halfs
  _Float16* sH1    = sBig + 16 * SPAD;      // 16 x HID halfs
  float*    logitsF= (float*)sBig;          // 16 x 16 f32
  float*    gatesF = (float*)sBig;          // 16 x 256 f32
  _Float16* sP     = sBig;                  // 80 x SPAD halfs (pred_step)
  _Float16* sQKV   = sBig;                  // 80 x QKVD halfs
  _Float16* sAO    = sBig + 80 * QKVD;      // 80 x HID halfs (attention out)

  // ---- load + pad input tile to f16
  for (int idx = lane; idx < 16 * SPAD; idx += 32) {
    int r = idx / SPAD, c = idx - r * SPAD;
    int cc = (c < STATE) ? c : 0;           // clamp, select below
    float xv = p.x[(size_t)(b0 + r) * STATE + cc];
    sX[idx] = (c < STATE) ? (_Float16)xv : (_Float16)0.0f;
  }
  __syncthreads();

  // ---- fc1 (relu) -> sH1
  gemm_tile(sX, SPAD, p.fc1_w, STATE, SPAD, HID, lane, [&](int n0, const v8f& d) {
    int n = (lane & 15) + n0, hi = lane >> 4;
    float bv = p.fc1_b[n];
#pragma unroll
    for (int r = 0; r < 8; ++r)
      sH1[(r + hi * 8) * HID + n] = (_Float16)fmaxf(d[r] + bv, 0.0f);
  });
  __syncthreads();

  // ---- fc2 (relu) -> sH2
  gemm_tile(sH1, HID, p.fc2_w, HID, HID, HID, lane, [&](int n0, const v8f& d) {
    int n = (lane & 15) + n0, hi = lane >> 4;
    float bv = p.fc2_b[n];
#pragma unroll
    for (int r = 0; r < 8; ++r)
      sH2[(r + hi * 8) * HID + n] = (_Float16)fmaxf(d[r] + bv, 0.0f);
  });
  __syncthreads();

  // ---- fc3 logits (f32, scratch reuses sBig; sX/sH1 dead)
  gemm_tile(sH2, HID, p.fc3_w, HID, HID, ACT, lane, [&](int n0, const v8f& d) {
    int n = (lane & 15) + n0, hi = lane >> 4;
    if (n < ACT) {
      float bv = p.fc3_b[n];
#pragma unroll
      for (int r = 0; r < 8; ++r) logitsF[(r + hi * 8) * 16 + n] = d[r] + bv;
    }
  });
  __syncthreads();

  // ---- action softmax -> global
  if (lane < 16) {
    int r = lane;
    float mx = -1e30f;
    for (int j = 0; j < ACT; ++j) mx = fmaxf(mx, logitsF[r * 16 + j]);
    float sum = 0.0f;
    for (int j = 0; j < ACT; ++j) sum += __expf(logitsF[r * 16 + j] - mx);
    float inv = 1.0f / sum;
    for (int j = 0; j < ACT; ++j)
      p.out_probs[(size_t)(b0 + r) * ACT + j] = __expf(logitsF[r * 16 + j] - mx) * inv;
  }
  __syncthreads();

  // ---- h0 / c0 -> sH (f16), sC (f32)
  gemm_tile(sH2, HID, p.h0_w, HID, HID, HID, lane, [&](int n0, const v8f& d) {
    int n = (lane & 15) + n0, hi = lane >> 4;
    float bv = p.h0_b[n];
#pragma unroll
    for (int r = 0; r < 8; ++r) sH[(r + hi * 8) * HID + n] = (_Float16)(d[r] + bv);
  });
  gemm_tile(sH2, HID, p.c0_w, HID, HID, HID, lane, [&](int n0, const v8f& d) {
    int n = (lane & 15) + n0, hi = lane >> 4;
    float bv = p.c0_b[n];
#pragma unroll
    for (int r = 0; r < 8; ++r) sC[(r + hi * 8) * HID + n] = d[r] + bv;
  });
  __syncthreads();

  // ================= prediction steps =================
  for (int s = 0; s < STEPS; ++s) {
    // gates = h @ whh^T + (bih + bhh)   (wih * zeros dropped)
    gemm_tile(sH, HID, p.lstm_whh, HID, HID, NGATES, lane, [&](int n0, const v8f& d) {
      int n = (lane & 15) + n0, hi = lane >> 4;
      float bv = p.lstm_bih[n] + p.lstm_bhh[n];
#pragma unroll
      for (int r = 0; r < 8; ++r) gatesF[(r + hi * 8) * NGATES + n] = d[r] + bv;
    });
    __syncthreads();

    // LSTM elementwise update
    for (int idx = lane; idx < 16 * HID; idx += 32) {
      int r = idx >> 6, c = idx & 63;
      float gi = gatesF[r * NGATES + c];
      float gf = gatesF[r * NGATES + 64 + c];
      float gg = gatesF[r * NGATES + 128 + c];
      float go = gatesF[r * NGATES + 192 + c];
      float cn = sigm(gf) * sC[idx] + sigm(gi) * tanhf(gg);
      sC[idx] = cn;
      sH[idx] = (_Float16)(sigm(go) * tanhf(cn));
    }
    __syncthreads();

    // zero K-pad columns of pred_step (gates scratch overwrote them)
    for (int idx = lane; idx < 80 * (SPAD - STATE); idx += 32) {
      int r = idx / (SPAD - STATE), c = STATE + idx - r * (SPAD - STATE);
      sP[r * SPAD + c] = (_Float16)0.0f;
    }
    __syncthreads();

    // predfc: [16 x 750] reshaped into 80 rows (b*TEAM+t) x SPAD
    gemm_tile(sH, HID, p.predfc_w, HID, HID, TEAM * STATE, lane, [&](int n0, const v8f& d) {
      int n = (lane & 15) + n0, hi = lane >> 4;
      if (n < TEAM * STATE) {
        float bv = p.predfc_b[n];
        int t = n / STATE, c = n - t * STATE;
#pragma unroll
        for (int r = 0; r < 8; ++r) {
          int m = r + hi * 8;
          sP[(m * TEAM + t) * SPAD + c] = (_Float16)(d[r] + bv);
        }
      }
    });
    __syncthreads();

    // embeds: five 16-row M tiles over the 80-row matrix
    for (int t = 0; t < TEAM; ++t) {
      gemm_tile(sP + t * 16 * SPAD, SPAD, p.emb_w, STATE, SPAD, HID, lane,
                [&](int n0, const v8f& d) {
        int n = (lane & 15) + n0, hi = lane >> 4;
        float bv = p.emb_b[n];
#pragma unroll
        for (int r = 0; r < 8; ++r)
          sE[(t * 16 + r + hi * 8) * HID + n] = (_Float16)(d[r] + bv);
      });
    }
    __syncthreads();

    // qkv projection (pred_step scratch dead -> reuse sBig)
    for (int t = 0; t < TEAM; ++t) {
      gemm_tile(sE + t * 16 * HID, HID, p.attn_in_w, HID, HID, QKVD, lane,
                [&](int n0, const v8f& d) {
        int n = (lane & 15) + n0, hi = lane >> 4;
        float bv = p.attn_in_b[n];
#pragma unroll
        for (int r = 0; r < 8; ++r)
          sQKV[(t * 16 + r + hi * 8) * QKVD + n] = (_Float16)(d[r] + bv);
      });
    }
    __syncthreads();

    // attention: one (batch b, head h) pair per lane; S=5, D=16, scale 1/sqrt(16)
    for (int it = 0; it < 2; ++it) {
      int pidx = lane + 32 * it;            // 0..63
      int b = pidx >> 2, h = pidx & 3;
      float sc[TEAM][TEAM];
#pragma unroll
      for (int i = 0; i < TEAM; ++i) {
        const _Float16* qi = sQKV + (b * TEAM + i) * QKVD + h * 16;
#pragma unroll
        for (int j = 0; j < TEAM; ++j) {
          const _Float16* kj = sQKV + (b * TEAM + j) * QKVD + 64 + h * 16;
          float acc = 0.0f;
#pragma unroll
          for (int d2 = 0; d2 < 16; ++d2) acc += (float)qi[d2] * (float)kj[d2];
          sc[i][j] = acc * 0.25f;
        }
      }
#pragma unroll
      for (int i = 0; i < TEAM; ++i) {
        float mx = sc[i][0];
#pragma unroll
        for (int j = 1; j < TEAM; ++j) mx = fmaxf(mx, sc[i][j]);
        float sum = 0.0f;
#pragma unroll
        for (int j = 0; j < TEAM; ++j) { sc[i][j] = __expf(sc[i][j] - mx); sum += sc[i][j]; }
        float inv = 1.0f / sum;
#pragma unroll
        for (int j = 0; j < TEAM; ++j) sc[i][j] *= inv;
      }
#pragma unroll
      for (int i = 0; i < TEAM; ++i) {
#pragma unroll
        for (int d2 = 0; d2 < 16; ++d2) {
          float acc = 0.0f;
#pragma unroll
          for (int j = 0; j < TEAM; ++j)
            acc += sc[i][j] * (float)sQKV[(b * TEAM + j) * QKVD + 128 + h * 16 + d2];
          sAO[(b * TEAM + i) * HID + h * 16 + d2] = (_Float16)acc;
        }
      }
    }
    __syncthreads();

    // output projection -> sE (embeds dead)
    for (int t = 0; t < TEAM; ++t) {
      gemm_tile(sAO + t * 16 * HID, HID, p.attn_out_w, HID, HID, HID, lane,
                [&](int n0, const v8f& d) {
        int n = (lane & 15) + n0, hi = lane >> 4;
        float bv = p.attn_out_b[n];
#pragma unroll
        for (int r = 0; r < 8; ++r)
          sE[(t * 16 + r + hi * 8) * HID + n] = (_Float16)(d[r] + bv);
      });
    }
    __syncthreads();

    // unembed -> global pred [B, STEPS, TEAM, STATE]
    for (int t = 0; t < TEAM; ++t) {
      gemm_tile(sE + t * 16 * HID, HID, p.unemb_w, HID, HID, STATE, lane,
                [&](int n0, const v8f& d) {
        int n = (lane & 15) + n0, hi = lane >> 4;
        if (n < STATE) {
          float bv = p.unemb_b[n];
#pragma unroll
          for (int r = 0; r < 8; ++r) {
            int rg = t * 16 + r + hi * 8;          // 0..79 = b*TEAM+team
            int b = rg / TEAM, tm = rg - b * TEAM;
            size_t o = ((((size_t)(b0 + b) * STEPS + s) * TEAM + tm) * STATE) + n;
            p.out_pred[o] = d[r] + bv;
          }
        }
      });
    }
    __syncthreads();
  }
}

extern "C" void kernel_launch(void* const* d_in, const int* in_sizes, int n_in,
                              void* d_out, int out_size, void* d_ws, size_t ws_size,
                              hipStream_t stream) {
  (void)in_sizes; (void)n_in; (void)out_size; (void)d_ws; (void)ws_size;
  Params p;
  p.x         = (const float*)d_in[0];
  p.fc1_w     = (const float*)d_in[1];  p.fc1_b     = (const float*)d_in[2];
  p.fc2_w     = (const float*)d_in[3];  p.fc2_b     = (const float*)d_in[4];
  p.fc3_w     = (const float*)d_in[5];  p.fc3_b     = (const float*)d_in[6];
  p.h0_w      = (const float*)d_in[7];  p.h0_b      = (const float*)d_in[8];
  p.c0_w      = (const float*)d_in[9];  p.c0_b      = (const float*)d_in[10];
  /* d_in[11] = lstm_wih: multiplied by a zero input, mathematically dead */
  p.lstm_whh  = (const float*)d_in[12];
  p.lstm_bih  = (const float*)d_in[13]; p.lstm_bhh  = (const float*)d_in[14];
  p.predfc_w  = (const float*)d_in[15]; p.predfc_b  = (const float*)d_in[16];
  p.emb_w     = (const float*)d_in[17]; p.emb_b     = (const float*)d_in[18];
  p.attn_in_w = (const float*)d_in[19]; p.attn_in_b = (const float*)d_in[20];
  p.attn_out_w= (const float*)d_in[21]; p.attn_out_b= (const float*)d_in[22];
  p.unemb_w   = (const float*)d_in[23]; p.unemb_b   = (const float*)d_in[24];
  p.out_probs = (float*)d_out;
  p.out_pred  = (float*)d_out + (size_t)BATCH * ACT;

  policy_fused<<<dim3(BATCH / 16), dim3(32), 0, stream>>>(p);
}
